// RCAM_68161130987577
// MI455X (gfx1250) — matrix operand order repaired and checked by reading, not compiled, not run on hardware
//
#include <hip/hip_runtime.h>
#include <hip/hip_bf16.h>

typedef __bf16 bf16;
typedef __attribute__((ext_vector_type(16))) __bf16 v16bf;
typedef __attribute__((ext_vector_type(8)))  float  v8f;

#define LQ 2048
#define LK 2048
#define BN 4
#define DM 256
#define MH 32
#define HH 8
#define HD 2048  // H*D

// padded LDS strides (conflict-free, rows stay 16B aligned)
#define LD256 264   // for 256-wide bf16 tiles: 528B row stride
#define LD32  40    // for 32-wide bf16 tiles:  80B row stride

#if defined(__has_builtin)
#if __has_builtin(__builtin_amdgcn_global_load_async_to_lds_b128) && \
    __has_builtin(__builtin_amdgcn_s_wait_asynccnt)
#define HAVE_ASYNC_LDS 1
#endif
#endif

// async-LDS builtin wants pointers to 16-byte int vectors
typedef int v4i_vs __attribute__((vector_size(16)));
typedef __attribute__((address_space(1))) v4i_vs gv4i;
typedef __attribute__((address_space(3))) v4i_vs lv4i;

// ---------------- WMMA helpers (CDNA5 wave32 layouts, ISA 7.12.2) ----------------

__device__ __forceinline__ v8f wmma_bf16(v16bf a, v16bf b, v8f c) {
  return __builtin_amdgcn_wmma_f32_16x16x32_bf16(false, a, false, b, (short)0, c,
                                                 false, false);
}

// A fragment (16x32 bf16) from row-major matrix, row stride ld, base at (m0,k0).
// lanes 0-15: row M=lane, V0..V3 K=0..7, V4..V7 K=16..23; lanes 16-31: K+8.
__device__ __forceinline__ v16bf frag_A_rm(const bf16* base, int ld) {
  const int lane = threadIdx.x & 31;
  const int half = lane >> 4;
  const bf16* rp = base + (lane & 15) * ld;
  v16bf a;
#pragma unroll
  for (int r = 0; r < 8; ++r) {
    const int kb = ((r < 4) ? (2 * r) : (2 * r + 8)) + 8 * half;
    a[2 * r]     = rp[kb];
    a[2 * r + 1] = rp[kb + 1];
  }
  return a;
}

// B fragment (32x16, K x N) where source is stored [N][K] row-major (B = src^T).
// lanes 0-15: col N=lane, V0..V3 K=0..7, V4..V7 K=16..23; lanes 16-31: K+8.
__device__ __forceinline__ v16bf frag_B_nk(const bf16* base, int ld) {
  const int lane = threadIdx.x & 31;
  const int half = lane >> 4;
  const bf16* cp = base + (lane & 15) * ld;
  v16bf b;
#pragma unroll
  for (int r = 0; r < 8; ++r) {
    const int kb = ((r < 4) ? (2 * r) : (2 * r + 8)) + 8 * half;
    b[2 * r]     = cp[kb];
    b[2 * r + 1] = cp[kb + 1];
  }
  return b;
}

// ---------------- Kernel 1: q0/k0/v0 = input + pos, f32 -> bf16, [L,B,D]->[B,L,D] ----
__global__ void k_prep(const float* __restrict__ q, const float* __restrict__ k,
                       const float* __restrict__ v, const float* __restrict__ p1,
                       const float* __restrict__ p2, bf16* __restrict__ q0,
                       bf16* __restrict__ k0, bf16* __restrict__ v0) {
  const int idx = blockIdx.x * blockDim.x + threadIdx.x;  // one per 8 elems
  const int TOT = BN * LQ * (DM / 8);
  if (idx >= TOT) return;
  const int d8 = idx & 31;
  const int t  = idx >> 5;
  const int l  = t & (LQ - 1);
  const int b  = t >> 11;
  const int in  = ((l * BN + b) << 8) + (d8 << 3);
  const int out = ((b * LQ + l) << 8) + (d8 << 3);
#pragma unroll
  for (int i = 0; i < 8; ++i) {
    q0[out + i] = (bf16)(q[in + i] + p1[in + i]);
    k0[out + i] = (bf16)(k[in + i] + p2[in + i]);
    v0[out + i] = (bf16)(v[in + i] + p2[in + i]);
  }
}

// ---------------- Kernel 2: f32 -> bf16 weight convert ----------------
__global__ void k_cvt(const float* __restrict__ src, bf16* __restrict__ dst, int n) {
  const int i = blockIdx.x * blockDim.x + threadIdx.x;
  if (i < n) dst[i] = (bf16)src[i];
}

// ---------------- Kernel 3: qh/kh = l2norm(X @ W^T), W: [H][M][D] ----------------
// grid: H*B*(LQ/128); block 256 (8 waves, 16 rows each)
__global__ void k_proj_qk(const bf16* __restrict__ X, const bf16* __restrict__ W,
                          bf16* __restrict__ Y) {
  __shared__ bf16 s_w[MH * LD256];       // padded 32x256 tile
  __shared__ float s_t[8][16][MH];
  const int rb = blockIdx.x & 15;
  const int hb = blockIdx.x >> 4;        // h*B + b
  const int h  = hb >> 2;
  const int b  = hb & 3;
  const bf16* wsrc = W + h * MH * DM;
  for (int i = threadIdx.x; i < MH * (DM / 8); i += blockDim.x) {
    const int r = i >> 5, c = i & 31;
    ((uint4*)(s_w + r * LD256))[c] = ((const uint4*)(wsrc + r * DM))[c];
  }
  __syncthreads();
  const int w  = threadIdx.x >> 5;
  const int l0 = rb * 128 + w * 16;
  const bf16* Xb = X + ((size_t)b * LQ + l0) * DM;
  v8f acc0 = {}, acc1 = {};
#pragma unroll
  for (int s = 0; s < 8; ++s) {
    v16bf a  = frag_A_rm(Xb + 32 * s, DM);
    v16bf b0 = frag_B_nk(s_w + 32 * s, LD256);
    v16bf b1 = frag_B_nk(s_w + 16 * LD256 + 32 * s, LD256);
    acc0 = wmma_bf16(a, b0, acc0);
    acc1 = wmma_bf16(a, b1, acc1);
  }
  const int lane = threadIdx.x & 31;
  const int half = lane >> 4;
  const int n = lane & 15;
#pragma unroll
  for (int jj = 0; jj < 8; ++jj) {
    const int m = jj + 8 * half;
    s_t[w][m][n]      = acc0[jj];
    s_t[w][m][16 + n] = acc1[jj];
  }
  __syncthreads();
  if (threadIdx.x < 128) {
    const int w2 = threadIdx.x >> 4, m = threadIdx.x & 15;
    float ss = 0.f;
#pragma unroll
    for (int c = 0; c < MH; ++c) { float x = s_t[w2][m][c]; ss += x * x; }
    const float inv = 1.f / fmaxf(sqrtf(ss), 1e-12f);
    const int l = rb * 128 + w2 * 16 + m;
    bf16* dst = Y + ((size_t)hb * LQ + l) * MH;
#pragma unroll
    for (int c = 0; c < MH; ++c) dst[c] = (bf16)(s_t[w2][m][c] * inv);
  }
}

// ---------------- Kernel 4: vh = v0 @ WV^T per (h,b) ----------------
// grid: H*B*(LK/64); block 256; dyn LDS: WV[h] padded 135 KB
__global__ void k_proj_v(const bf16* __restrict__ V0, const bf16* __restrict__ WV,
                         bf16* __restrict__ VH) {
  extern __shared__ __align__(16) char smem[];
  bf16* s_w = (bf16*)smem;               // 256 x LD256
  const int rb = blockIdx.x & 31;
  const int hb = blockIdx.x >> 5;
  const int h  = hb >> 2;
  const int b  = hb & 3;
  const bf16* wsrc = WV + (size_t)h * DM * DM;
  for (int i = threadIdx.x; i < DM * (DM / 8); i += blockDim.x) {
    const int r = i >> 5, c = i & 31;
    ((uint4*)(s_w + r * LD256))[c] = ((const uint4*)(wsrc + r * DM))[c];
  }
  __syncthreads();
  const int w  = threadIdx.x >> 5;
  const int rt = w & 3;
  const int cg = w >> 2;
  const int r0 = rb * 64 + rt * 16;
  const bf16* A = V0 + ((size_t)b * LK + r0) * DM;
  v16bf afr[8];
#pragma unroll
  for (int s = 0; s < 8; ++s) afr[s] = frag_A_rm(A + 32 * s, DM);
  const int lane = threadIdx.x & 31;
  const int half = lane >> 4;
  const int n = lane & 15;
#pragma unroll
  for (int j = 0; j < 8; ++j) {
    const int n0 = (cg * 8 + j) * 16;
    v8f acc = {};
#pragma unroll
    for (int s = 0; s < 8; ++s)
      acc = wmma_bf16(afr[s], frag_B_nk(s_w + n0 * LD256 + 32 * s, LD256), acc);
#pragma unroll
    for (int jj = 0; jj < 8; ++jj) {
      const int m = jj + 8 * half;
      VH[((size_t)hb * LK + r0 + m) * DM + n0 + n] = (bf16)acc[jj];
    }
  }
}

// ---------------- Kernel 5: scores + row softmax + column-sum atomics ----------------
// grid: H*B*(LQ/32); block 256; dyn LDS: S[32][2056] f32 (~257 KB) + reductions
#define SLD 2056
__global__ void k_scores(const bf16* __restrict__ QH, const bf16* __restrict__ KH,
                         bf16* __restrict__ P, float* __restrict__ colsum) {
  extern __shared__ __align__(16) char smem[];
  float* S    = (float*)smem;            // 32 * SLD
  float* red  = S + 32 * SLD;            // 256
  float* rinv = red + 256;               // 32
  const int rb = blockIdx.x & 63;
  const int hb = blockIdx.x >> 6;
  const int l0 = rb * 32;
  const int w  = threadIdx.x >> 5;
  const int lane = threadIdx.x & 31;
  const int half = lane >> 4;
  const int n = lane & 15;

  const bf16* qbase = QH + ((size_t)hb * LQ + l0) * MH;
  v16bf a0 = frag_A_rm(qbase, MH);
  v16bf a1 = frag_A_rm(qbase + 16 * MH, MH);
#pragma unroll 1
  for (int i = 0; i < 16; ++i) {
    const int ct = w + 8 * i;            // 128 column tiles of 16
    v16bf bb = frag_B_nk(KH + ((size_t)hb * LK + ct * 16) * MH, MH);
    v8f c0 = {}, c1 = {};
    c0 = wmma_bf16(a0, bb, c0);
    c1 = wmma_bf16(a1, bb, c1);
#pragma unroll
    for (int jj = 0; jj < 8; ++jj) {
      const int m = jj + 8 * half;
      S[m * SLD + ct * 16 + n]        = c0[jj];
      S[(16 + m) * SLD + ct * 16 + n] = c1[jj];
    }
  }
  __syncthreads();

  // softmax: 8 threads per row, 256-col chunks
  const int row = threadIdx.x >> 3;
  const int ch  = threadIdx.x & 7;
  const int cb  = ch * 256;
  float mx = -3.4e38f;
  for (int i = 0; i < 256; ++i) mx = fmaxf(mx, S[row * SLD + cb + i]);
  red[row * 8 + ch] = mx;
  __syncthreads();
  float rm = red[row * 8];
#pragma unroll
  for (int j = 1; j < 8; ++j) rm = fmaxf(rm, red[row * 8 + j]);
  __syncthreads();
  float sum = 0.f;
  for (int i = 0; i < 256; ++i) {
    float e = __expf(S[row * SLD + cb + i] - rm);
    S[row * SLD + cb + i] = e;
    sum += e;
  }
  red[row * 8 + ch] = sum;
  __syncthreads();
  float rs = 0.f;
#pragma unroll
  for (int j = 0; j < 8; ++j) rs += red[row * 8 + j];
  const float inv = 1.f / rs;
  if (ch == 0) rinv[row] = inv;
  bf16* pdst = P + ((size_t)hb * LQ + l0 + row) * LK + cb;
  for (int i = 0; i < 256; ++i) pdst[i] = (bf16)(S[row * SLD + cb + i] * inv);
  __syncthreads();

  const int c = threadIdx.x * 8;
#pragma unroll 1
  for (int cc = c; cc < c + 8; ++cc) {
    float acc = 0.f;
#pragma unroll
    for (int r = 0; r < 32; ++r) acc += S[r * SLD + cc] * rinv[r];
    atomicAdd(&colsum[(size_t)hb * LK + cc], acc);
  }
}

// ---------------- Kernel 6: fused y = (P/(T+colsum)) @ vh ; z = relu((q0-y)@WL^T) ----
// grid: H*B*(LQ/32); block 256
// LDS: WL (padded) + P chunk 32x32 (padded) + vh chunk transposed [e][k] + X tile
__global__ void k_yz(const bf16* __restrict__ P, const bf16* __restrict__ VH,
                     const bf16* __restrict__ WL, const float* __restrict__ cs,
                     const float* __restrict__ q, const float* __restrict__ p1,
                     float* __restrict__ out) {
  extern __shared__ __align__(16) char smem[];
  bf16* s_wl = (bf16*)smem;              // DM * LD256
  bf16* s_p  = s_wl + DM * LD256;        // 32 * LD32
  bf16* s_vT = s_p + 32 * LD32;          // DM * LD32 (transposed: [e][k])
  bf16* s_x  = s_vT + DM * LD32;         // 32 * LD256
  const int rb = blockIdx.x & 63;
  const int hb = blockIdx.x >> 6;
  const int h  = hb >> 2;
  const int b  = hb & 3;
  const int l0 = rb * 32;
  {
    const bf16* wsrc = WL + (size_t)h * DM * DM;
    for (int i = threadIdx.x; i < DM * (DM / 8); i += blockDim.x) {
      const int r = i >> 5, c = i & 31;
      ((uint4*)(s_wl + r * LD256))[c] = ((const uint4*)(wsrc + r * DM))[c];
    }
  }

  const int w  = threadIdx.x >> 5;
  const int rt = w & 1;
  const int cbase = (w >> 1) * 64;
  const int lane = threadIdx.x & 31;
  const int half = lane >> 4;
  const int n = lane & 15;

  const bf16* Pbase = P + ((size_t)hb * LQ + l0) * LK;
  const bf16* Vbase = VH + (size_t)hb * LK * DM;
  const float* csb  = cs + (size_t)hb * LK;
  v8f acc[4] = {{}, {}, {}, {}};

#pragma unroll 1
  for (int s = 0; s < 64; ++s) {         // K chunks of 32 over Lk
    const int k0 = s * 32;
    // --- stage P chunk 32x32 (async global->LDS when available) ---
#if defined(HAVE_ASYNC_LDS)
    if (threadIdx.x < 128) {
      const int r = threadIdx.x >> 2, c8 = threadIdx.x & 3;
      __builtin_amdgcn_global_load_async_to_lds_b128(
          (gv4i*)(Pbase + (size_t)r * LK + k0 + c8 * 8),
          (lv4i*)(s_p + r * LD32 + c8 * 8), 0, 0);
    }
#else
    if (threadIdx.x < 128) {
      const int r = threadIdx.x >> 2, c8 = threadIdx.x & 3;
      ((uint4*)(s_p + r * LD32))[c8] =
          *(const uint4*)(Pbase + (size_t)r * LK + k0 + c8 * 8);
    }
#endif
    // --- stage vh chunk transposed, folding in column normalization ---
    for (int i = threadIdx.x; i < 1024; i += 256) {
      const int r = i >> 5, c8 = i & 31;   // r = k row, c8*8 = e base
      const uint4 u = *(const uint4*)(Vbase + (size_t)(k0 + r) * DM + c8 * 8);
      const float sc = 1.f / (1e-9f + csb[k0 + r]);
      const bf16* pv = (const bf16*)&u;
#pragma unroll
      for (int ii = 0; ii < 8; ++ii)
        s_vT[(c8 * 8 + ii) * LD32 + r] = (bf16)((float)pv[ii] * sc);
    }
    // prefetch next chunk while this one computes
    if (s < 63) {
      const int kn = k0 + 32;
      if (threadIdx.x < 128)
        __builtin_prefetch(
            Pbase + (size_t)(threadIdx.x >> 2) * LK + kn + (threadIdx.x & 3) * 8,
            0, 0);
      __builtin_prefetch(Vbase + (size_t)(kn + (threadIdx.x & 31)) * DM, 0, 0);
    }
#if defined(HAVE_ASYNC_LDS)
    __builtin_amdgcn_s_wait_asynccnt(0);
#endif
    __syncthreads();
    v16bf a = frag_A_rm(s_p + rt * 16 * LD32, LD32);
#pragma unroll
    for (int j = 0; j < 4; ++j)
      acc[j] = wmma_bf16(a, frag_B_nk(s_vT + (cbase + j * 16) * LD32, LD32),
                         acc[j]);
    __syncthreads();
  }

  // X = q0 - y  (recompute q0 from f32 inputs), store bf16 to LDS
#pragma unroll
  for (int j = 0; j < 4; ++j) {
    const int n0 = cbase + j * 16;
#pragma unroll
    for (int jj = 0; jj < 8; ++jj) {
      const int m = jj + 8 * half;
      const int l = l0 + rt * 16 + m;
      const int e = n0 + n;
      const float qv = q[((size_t)l * BN + b) * DM + e] +
                       p1[((size_t)l * BN + b) * DM + e];
      s_x[(rt * 16 + m) * LD256 + e] = (bf16)(qv - acc[j][jj]);
    }
  }
  __syncthreads();

  // z = relu(X @ WL^T)
  v8f z4[4] = {{}, {}, {}, {}};
#pragma unroll
  for (int t = 0; t < 8; ++t) {
    v16bf a = frag_A_rm(s_x + rt * 16 * LD256 + 32 * t, LD256);
#pragma unroll
    for (int j = 0; j < 4; ++j)
      z4[j] = wmma_bf16(
          a, frag_B_nk(s_wl + (cbase + j * 16) * LD256 + 32 * t, LD256), z4[j]);
  }
#pragma unroll
  for (int j = 0; j < 4; ++j) {
    const int n0 = cbase + j * 16;
#pragma unroll
    for (int jj = 0; jj < 8; ++jj) {
      const int m = jj + 8 * half;
      const int l = l0 + rt * 16 + m;
      out[((size_t)l * BN + b) * HD + h * DM + n0 + n] = fmaxf(z4[j][jj], 0.f);
    }
  }
}

// ---------------- host launch ----------------
extern "C" void kernel_launch(void* const* d_in, const int* in_sizes, int n_in,
                              void* d_out, int out_size, void* d_ws, size_t ws_size,
                              hipStream_t stream) {
  const float* q  = (const float*)d_in[0];
  const float* k  = (const float*)d_in[1];
  const float* v  = (const float*)d_in[2];
  const float* p1 = (const float*)d_in[3];
  const float* p2 = (const float*)d_in[4];
  const float* WK = (const float*)d_in[5];
  const float* WQ = (const float*)d_in[6];
  const float* WV = (const float*)d_in[7];
  const float* WL = (const float*)d_in[8];
  float* out = (float*)d_out;

  char* ws = (char*)d_ws;
  size_t off = 0;
  auto alloc = [&](size_t bytes) {
    char* p = ws + off;
    off = (off + bytes + 255) & ~(size_t)255;
    return p;
  };
  bf16* q0b = (bf16*)alloc((size_t)BN * LQ * DM * 2);
  bf16* k0b = (bf16*)alloc((size_t)BN * LK * DM * 2);
  bf16* v0b = (bf16*)alloc((size_t)BN * LK * DM * 2);
  bf16* wqb = (bf16*)alloc((size_t)HH * MH * DM * 2);
  bf16* wkb = (bf16*)alloc((size_t)HH * MH * DM * 2);
  bf16* wvb = (bf16*)alloc((size_t)HH * DM * DM * 2);
  bf16* wlb = (bf16*)alloc((size_t)HH * DM * DM * 2);
  bf16* qh  = (bf16*)alloc((size_t)HH * BN * LQ * MH * 2);
  bf16* kh  = (bf16*)alloc((size_t)HH * BN * LK * MH * 2);
  bf16* vh  = (bf16*)alloc((size_t)HH * BN * LK * DM * 2);
  float* cs = (float*)alloc((size_t)HH * BN * LK * 4);
  bf16* Pm  = (bf16*)alloc((size_t)HH * BN * LQ * LK * 2);

  // 1. elementwise prep + weight conversion
  k_prep<<<(BN * LQ * DM / 8 + 255) / 256, 256, 0, stream>>>(q, k, v, p1, p2, q0b,
                                                             k0b, v0b);
  k_cvt<<<(HH * MH * DM + 255) / 256, 256, 0, stream>>>(WQ, wqb, HH * MH * DM);
  k_cvt<<<(HH * MH * DM + 255) / 256, 256, 0, stream>>>(WK, wkb, HH * MH * DM);
  k_cvt<<<(HH * DM * DM + 255) / 256, 256, 0, stream>>>(WV, wvb, HH * DM * DM);
  k_cvt<<<(HH * DM * DM + 255) / 256, 256, 0, stream>>>(WL, wlb, HH * DM * DM);

  // 2. projections
  k_proj_qk<<<HH * BN * (LQ / 128), 256, 0, stream>>>(q0b, wqb, qh);
  k_proj_qk<<<HH * BN * (LK / 128), 256, 0, stream>>>(k0b, wkb, kh);
  k_proj_v<<<HH * BN * (LK / 64), 256, (size_t)DM * LD256 * 2, stream>>>(v0b, wvb,
                                                                         vh);

  // 3. scores + softmax + column sums
  (void)hipMemsetAsync(cs, 0, (size_t)HH * BN * LK * 4, stream);
  const size_t smem_s = (32 * SLD + 256 + 32) * sizeof(float);
  k_scores<<<HH * BN * (LQ / 32), 256, smem_s, stream>>>(qh, kh, Pm, cs);

  // 4. fused: column normalization folded into vh staging, y + output GEMM + ReLU
  const size_t smem_yz =
      (size_t)(DM * LD256 + 32 * LD32 + DM * LD32 + 32 * LD256) * 2;
  k_yz<<<HH * BN * (LQ / 32), 256, smem_yz, stream>>>(Pm, vh, wlb, cs, q, p1, out);
}
